// AxialAttention_635655160666
// MI455X (gfx1250) — matrix operand order repaired and checked by reading, not compiled, not run on hardware
//
#include <hip/hip_runtime.h>
#include <cmath>

// CDNA5 (gfx1250) axial attention: B'=512 batches, S=512, c=64, wave32 WMMA f16->f32.
// Each wave owns a 32-row Q tile (2 M-tiles sharing all K/V B-fragments).
typedef __attribute__((ext_vector_type(16))) _Float16 v16h;
typedef __attribute__((ext_vector_type(8)))  float    v8f;

#define WMMA_F16(a, b, c) \
  __builtin_amdgcn_wmma_f32_16x16x32_f16(false, (a), false, (b), (short)0, (c), false, false)

#define ROWSTRIDE 512           // W*c = 8*64 floats between consecutive seq rows
#define SEQ       512
#define PSTRIDE   34            // padded f16 row stride for P staging (bank-conflict free)
#define NEGB      (-1.0e30f)    // finite "-inf" sentinel: exp2(NEGB - m) == 0, branchless
#define SCL2      0.1803368801111244f   // (1/sqrt(64)) * log2(e): softmax in exp2 domain

// B-fragment (32x16, f16) for Q*K^T: B[kc][key] = K[key0+n][cb + hi*8 + h + (h & 8)]
static __device__ __forceinline__ v16h load_b_k(const float* __restrict__ kb,
                                                int key0, int cb, int n, int hi) {
  const float* src = kb + (size_t)(key0 + n) * ROWSTRIDE + cb + hi * 8;
  v16h b;
#pragma unroll
  for (int h = 0; h < 16; ++h) b[h] = (_Float16)src[h + (h & 8)];
  return b;
}

// B-fragment (32x16, f16) for P*V: B[sv][col] = V[s0 + hi*8 + h + (h & 8)][nb + n]
static __device__ __forceinline__ v16h load_b_v(const float* __restrict__ vb,
                                                int s0, int nb, int n, int hi) {
  const float* src = vb + (size_t)(s0 + hi * 8) * ROWSTRIDE + nb + n;
  v16h b;
#pragma unroll
  for (int h = 0; h < 16; ++h) b[h] = (_Float16)src[(size_t)(h + (h & 8)) * ROWSTRIDE];
  return b;
}

__global__ __launch_bounds__(256) void axial_attn_wmma(
    const float* __restrict__ q, const float* __restrict__ k,
    const float* __restrict__ v, const int* __restrict__ amask,
    const float* __restrict__ hmask, float* __restrict__ out,
    float* __restrict__ attn) {
  __shared__ _Float16 pstage[8][2][16 * PSTRIDE];   // per-wave, per-M-tile P (16x32 f16)

  const int wave = threadIdx.x >> 5;
  const int lane = threadIdx.x & 31;
  const int n    = lane & 15;    // N column (B/C layouts) or M row (A layout)
  const int hi   = lane >> 4;    // half-wave selector

  const int bp   = blockIdx.x >> 1;          // flattened batch 0..511
  const int qblk = blockIdx.x & 1;
  const int q0   = qblk * 256 + wave * 32;   // this wave's 32-row Q tile

  const size_t gbase = (size_t)(bp >> 3) * 262144 + (size_t)(bp & 7) * 64;
  const float* qb = q + gbase;
  const float* kb = k + gbase;
  const float* vb = v + gbase;
  float* ob = out + gbase;
  float* ab = attn + (size_t)bp * (SEQ * SEQ);

  // ---- Q A-fragments (two 16-row M-tiles x two 32-wide K chunks) ----
  v16h aq[2][2];
#pragma unroll
  for (int mt = 0; mt < 2; ++mt) {
    const float* qrow = qb + (size_t)(q0 + mt * 16 + n) * ROWSTRIDE;  // row m = lane&15
#pragma unroll
    for (int t = 0; t < 8; ++t) {
      aq[mt][0][t]     = (_Float16)qrow[hi * 8 + t];        // K = hi*8 + t
      aq[mt][0][t + 8] = (_Float16)qrow[16 + hi * 8 + t];   // K = 16 + hi*8 + t
      aq[mt][1][t]     = (_Float16)qrow[32 + hi * 8 + t];
      aq[mt][1][t + 8] = (_Float16)qrow[48 + hi * 8 + t];
    }
  }

  float rmax[2][8], rsum[2][8];
#pragma unroll
  for (int mt = 0; mt < 2; ++mt)
#pragma unroll
    for (int r = 0; r < 8; ++r) { rmax[mt][r] = NEGB; rsum[mt][r] = 0.f; }

  // ================= pass 1: branchless online softmax stats (exp2 domain) ===========
  for (int kt = 0; kt < 32; ++kt) {
    const int key0 = kt * 16;
    __builtin_prefetch(kb + (size_t)(key0 + 16 + n) * ROWSTRIDE, 0, 1);  // speculative
    v16h b0 = load_b_k(kb, key0, 0, n, hi);
    v16h b1 = load_b_k(kb, key0, 32, n, hi);
    int mv[2][8];
#pragma unroll
    for (int mt = 0; mt < 2; ++mt)
#pragma unroll
      for (int r = 0; r < 8; ++r)
        mv[mt][r] = amask[(q0 + mt * 16 + r + 8 * hi) * SEQ + key0 + n];
#pragma unroll
    for (int mt = 0; mt < 2; ++mt) {
      v8f acc = {};
      acc = WMMA_F16(aq[mt][0], b0, acc);
      acc = WMMA_F16(aq[mt][1], b1, acc);
#pragma unroll
      for (int r = 0; r < 8; ++r) {
        float x  = mv[mt][r] ? acc[r] * SCL2 : NEGB;      // v_cndmask, no branch
        float nm = fmaxf(rmax[mt][r], x);
        rsum[mt][r] = rsum[mt][r] * __builtin_amdgcn_exp2f(rmax[mt][r] - nm)
                    + __builtin_amdgcn_exp2f(x - nm);
        rmax[mt][r] = nm;
      }
    }
  }
  // combine across the 16 lanes spanning N (masks < 16 keep halves separate)
#pragma unroll
  for (int mt = 0; mt < 2; ++mt)
#pragma unroll
    for (int r = 0; r < 8; ++r) {
#pragma unroll
      for (int d = 1; d < 16; d <<= 1) {
        float om = __shfl_xor(rmax[mt][r], d, 32);
        float os = __shfl_xor(rsum[mt][r], d, 32);
        float nm = fmaxf(rmax[mt][r], om);
        rsum[mt][r] = rsum[mt][r] * __builtin_amdgcn_exp2f(rmax[mt][r] - nm)
                    + os * __builtin_amdgcn_exp2f(om - nm);
        rmax[mt][r] = nm;
      }
    }
  float rinv[2][8];
#pragma unroll
  for (int mt = 0; mt < 2; ++mt)
#pragma unroll
    for (int r = 0; r < 8; ++r) rinv[mt][r] = 1.0f / rsum[mt][r];

  // ========== pass 2: recompute scores, emit attn, accumulate P*V ==========
  v8f oacc[2][4] = {};

  for (int dt = 0; dt < 16; ++dt) {           // 32-key contraction chunk per step
#pragma unroll
    for (int sub = 0; sub < 2; ++sub) {
      const int key0 = dt * 32 + sub * 16;
      v16h b0 = load_b_k(kb, key0, 0, n, hi);
      v16h b1 = load_b_k(kb, key0, 32, n, hi);
      int   mv[2][8];
      float hm[2][8];
#pragma unroll
      for (int mt = 0; mt < 2; ++mt)
#pragma unroll
        for (int r = 0; r < 8; ++r) {
          const int idx = (q0 + mt * 16 + r + 8 * hi) * SEQ + key0 + n;
          mv[mt][r] = amask[idx];
          hm[mt][r] = hmask[idx];
        }
#pragma unroll
      for (int mt = 0; mt < 2; ++mt) {
        v8f acc = {};
        acc = WMMA_F16(aq[mt][0], b0, acc);
        acc = WMMA_F16(aq[mt][1], b1, acc);
#pragma unroll
        for (int r = 0; r < 8; ++r) {
          const int qi   = q0 + mt * 16 + r + 8 * hi;
          const int kcol = key0 + n;
          float e = __builtin_amdgcn_exp2f(acc[r] * SCL2 - rmax[mt][r]) * rinv[mt][r];
          float p = mv[mt][r] ? e : 0.f;
          p *= hm[mt][r];
          ab[(size_t)qi * SEQ + kcol] = p;                 // coalesced attn store
          pstage[wave][mt][(r + 8 * hi) * PSTRIDE + sub * 16 + n] = (_Float16)p;
        }
      }
    }
    // Rebuild staged P as A-fragments (per-wave LDS; DS ops in-order per wave).
    v16h ap[2];
#pragma unroll
    for (int mt = 0; mt < 2; ++mt) {
      const _Float16* pbuf = &pstage[wave][mt][0];
#pragma unroll
      for (int t = 0; t < 8; ++t) {
        ap[mt][t]     = pbuf[n * PSTRIDE + hi * 8 + t];
        ap[mt][t + 8] = pbuf[n * PSTRIDE + 16 + hi * 8 + t];
      }
    }
    const int s0 = dt * 32;
    __builtin_prefetch(vb + (size_t)(s0 + 32 + hi * 16 + n) * ROWSTRIDE, 0, 1);
#pragma unroll
    for (int nb = 0; nb < 4; ++nb) {
      v16h bv = load_b_v(vb, s0, nb * 16, n, hi);          // shared by both M-tiles
      oacc[0][nb] = WMMA_F16(ap[0], bv, oacc[0][nb]);
      oacc[1][nb] = WMMA_F16(ap[1], bv, oacc[1][nb]);
    }
  }

  // ---- write out (C layout: row = q0 + mt*16 + r + 8*hi, col = nb*16 + n) ----
#pragma unroll
  for (int mt = 0; mt < 2; ++mt)
#pragma unroll
    for (int r = 0; r < 8; ++r) {
      const size_t row = (size_t)(q0 + mt * 16 + r + 8 * hi) * ROWSTRIDE;
      ob[row + 0  + n] = oacc[mt][0][r];
      ob[row + 16 + n] = oacc[mt][1][r];
      ob[row + 32 + n] = oacc[mt][2][r];
      ob[row + 48 + n] = oacc[mt][3][r];
    }
}

extern "C" void kernel_launch(void* const* d_in, const int* in_sizes, int n_in,
                              void* d_out, int out_size, void* d_ws, size_t ws_size,
                              hipStream_t stream) {
  (void)in_sizes; (void)n_in; (void)out_size; (void)d_ws; (void)ws_size;
  const float* q     = (const float*)d_in[0];
  const float* k     = (const float*)d_in[1];
  const float* v     = (const float*)d_in[2];
  const int*   amask = (const int*)d_in[3];
  const float* hmask = (const float*)d_in[4];
  float* out  = (float*)d_out;
  float* attn = out + (size_t)16777216;   // out is 2*8*4*512*8*64; attn follows

  // 512 batches * 2 q-blocks of 256 rows; 256 threads = 8 waves, 32 q-rows/wave
  axial_attn_wmma<<<dim3(1024), dim3(256), 0, stream>>>(q, k, v, amask, hmask, out, attn);
}
